// LogicMessagePassingNetwork_11003706213177
// MI455X (gfx1250) — matrix operand order repaired and checked by reading, not compiled, not run on hardware
//
#include <hip/hip_runtime.h>
#include <hip/hip_bf16.h>

typedef __attribute__((ext_vector_type(16))) _Float16 v16h;
typedef __attribute__((ext_vector_type(2)))  _Float16 h2;
typedef __attribute__((ext_vector_type(2)))  __fp16   f16x2;
typedef __attribute__((ext_vector_type(8)))  float    v8f;

union V16 {
  v16h  v;
  f16x2 p[8];
};

#define D1 64
#define D2 128

// ---------------- gather: x0 = fact_emb[edge_type] ----------------
__global__ __launch_bounds__(256) void gather_kernel(
    const float* __restrict__ fact, const int* __restrict__ etype,
    float* __restrict__ x0, int E) {
  int idx = blockIdx.x * blockDim.x + threadIdx.x;
  if (idx >= E * D1) return;
  int e = idx >> 6, k = idx & 63;
  x0[idx] = fact[etype[e] * D1 + k];
}

__global__ __launch_bounds__(256) void zero_kernel(float* __restrict__ p, int n) {
  int idx = blockIdx.x * blockDim.x + threadIdx.x;
  if (idx < n) p[idx] = 0.f;
}

// ---- triangle messages: agg[ac] += x[ab] * x[bc], lane-per-feature ----
__global__ __launch_bounds__(256) void tri_kernel(
    const float* __restrict__ x, const int* __restrict__ ab,
    const int* __restrict__ bc, const int* __restrict__ ac,
    float* __restrict__ agg, int T) {
  int idx = blockIdx.x * blockDim.x + threadIdx.x;
  int t = idx >> 6, k = idx & 63;
  if (t >= T) return;
  float v = x[ab[t] * D1 + k] * x[bc[t] * D1 + k];
  atomicAdd(&agg[ac[t] * D1 + k], v);
}

// Stage W[KD x ND] into LDS in fragment-major order: each lane's 16-half
// B fragment is contiguous (32B). frag = kstep*NT + nt.
// Fragment element pair j of lane l: k = kstep*32 + (l>>4)*8 + 2j + (j>=4 ? 8 : 0),
// n = nt*16 + (l&15)  (per the CDNA5 16-bit B operand layout).
template <int KD, int ND>
__device__ inline void stage_w_frag(const float* __restrict__ W, _Float16* Wl) {
  constexpr int NT = ND / 16;
  constexpr int NF = (KD / 32) * NT;
  for (int p = threadIdx.x; p < NF * 256; p += blockDim.x) {
    int f = p >> 8;
    int rem = p & 255;
    int l = rem >> 3;
    int j = rem & 7;
    int kstep = f / NT, nt = f % NT;
    int k = kstep * 32 + (l >> 4) * 8 + 2 * j + (j >= 4 ? 8 : 0);
    int n = nt * 16 + (l & 15);
    h2 v;
    v.x = (_Float16)W[k * ND + n];
    v.y = (_Float16)W[(k + 1) * ND + n];
    *(h2*)&Wl[(size_t)(f * 32 + l) * 16 + 2 * j] = v;
  }
}

// Build A fragment for one 16x32 K-step from a row-major f32 row.
// Per lane: two contiguous 8-float runs at (hi*8) and (hi*8 + 16).
__device__ inline v16h load_a_frag(const float* __restrict__ rowbase, int hi, bool valid) {
  float4 c0 = make_float4(0.f, 0.f, 0.f, 0.f), c1 = c0, c2 = c0, c3 = c0;
  if (valid) {
    const float* b = rowbase + hi * 8;
    c0 = *(const float4*)(b);
    c1 = *(const float4*)(b + 4);
    c2 = *(const float4*)(b + 16);
    c3 = *(const float4*)(b + 20);
  }
  V16 a;
  a.p[0] = __builtin_amdgcn_cvt_pkrtz(c0.x, c0.y);
  a.p[1] = __builtin_amdgcn_cvt_pkrtz(c0.z, c0.w);
  a.p[2] = __builtin_amdgcn_cvt_pkrtz(c1.x, c1.y);
  a.p[3] = __builtin_amdgcn_cvt_pkrtz(c1.z, c1.w);
  a.p[4] = __builtin_amdgcn_cvt_pkrtz(c2.x, c2.y);
  a.p[5] = __builtin_amdgcn_cvt_pkrtz(c2.z, c2.w);
  a.p[6] = __builtin_amdgcn_cvt_pkrtz(c3.x, c3.y);
  a.p[7] = __builtin_amdgcn_cvt_pkrtz(c3.z, c3.w);
  return a.v;
}

// ---- x_out = relu(agg @ W(64x64) + b); one 16-row tile per wave ----
__global__ __launch_bounds__(256) void layer_gemm(
    const float* __restrict__ A, const float* __restrict__ W,
    const float* __restrict__ bias, float* __restrict__ out, int E) {
  __shared__ _Float16 Wl[8 * 32 * 16];  // 2 ksteps x 4 ntiles, frag-major
  stage_w_frag<D1, D1>(W, Wl);
  __syncthreads();

  int lane = threadIdx.x & 31;
  int wave = threadIdx.x >> 5;
  int m = lane & 15, hi = lane >> 4;
  int rowbase = (blockIdx.x * 8 + wave) * 16;
  int arow = rowbase + m;
  bool valid = arow < E;

  const v16h* Wfrag = (const v16h*)Wl;

  v8f acc[4];
  v8f z = {0.f, 0.f, 0.f, 0.f, 0.f, 0.f, 0.f, 0.f};
#pragma unroll
  for (int nt = 0; nt < 4; ++nt) acc[nt] = z;

#pragma unroll
  for (int kstep = 0; kstep < 2; ++kstep) {
    v16h afrag = load_a_frag(A + (size_t)arow * D1 + kstep * 32, hi, valid);
#pragma unroll
    for (int nt = 0; nt < 4; ++nt) {
      v16h bfrag = Wfrag[(kstep * 4 + nt) * 32 + lane];
      acc[nt] = __builtin_amdgcn_wmma_f32_16x16x32_f16(
          false, afrag, false, bfrag, (short)0, acc[nt], false, false);
    }
  }

  // D layout: VGPR r -> row r + 8*hi; lane&15 -> col
#pragma unroll
  for (int nt = 0; nt < 4; ++nt) {
    int col = nt * 16 + m;
    float bv = bias[col];
#pragma unroll
    for (int r = 0; r < 8; ++r) {
      int row = rowbase + r + 8 * hi;
      if (row < E) {
        float v = acc[nt][r] + bv;
        out[(size_t)row * D1 + col] = v > 0.f ? v : 0.f;
      }
    }
  }
}

// ---- out = relu([x|x0] @ mlpW1 + b1) @ mlpW2 + b2, fully fused ----
__global__ __launch_bounds__(256) void final_mlp(
    const float* __restrict__ x, const float* __restrict__ x0,
    const float* __restrict__ W1, const float* __restrict__ b1,
    const float* __restrict__ W2, const float* __restrict__ b2,
    float* __restrict__ out, int E) {
  __shared__ _Float16 Wl[32 * 32 * 16];  // 4 ksteps x 8 ntiles, frag-major
  stage_w_frag<D2, D2>(W1, Wl);
  __syncthreads();

  int lane = threadIdx.x & 31;
  int wave = threadIdx.x >> 5;
  int m = lane & 15, hi = lane >> 4;
  int rowbase = (blockIdx.x * 8 + wave) * 16;
  int arow = rowbase + m;
  bool valid = arow < E;

  const v16h* Wfrag = (const v16h*)Wl;

  v8f acc[8];
  v8f z = {0.f, 0.f, 0.f, 0.f, 0.f, 0.f, 0.f, 0.f};
#pragma unroll
  for (int nt = 0; nt < 8; ++nt) acc[nt] = z;

#pragma unroll
  for (int kstep = 0; kstep < 4; ++kstep) {
    // feat = [x | x0]: ksteps 0-1 hit x, 2-3 hit x0 (clean 32-col split)
    const float* rowp = (kstep < 2) ? (x + (size_t)arow * D1 + kstep * 32)
                                    : (x0 + (size_t)arow * D1 + (kstep - 2) * 32);
    v16h afrag = load_a_frag(rowp, hi, valid);
#pragma unroll
    for (int nt = 0; nt < 8; ++nt) {
      v16h bfrag = Wfrag[(kstep * 8 + nt) * 32 + lane];
      acc[nt] = __builtin_amdgcn_wmma_f32_16x16x32_f16(
          false, afrag, false, bfrag, (short)0, acc[nt], false, false);
    }
  }

  // h = relu(acc + b1); fused GEMV with mlp_W2 (128 -> 1)
  float part[8];
#pragma unroll
  for (int r = 0; r < 8; ++r) part[r] = 0.f;
#pragma unroll
  for (int nt = 0; nt < 8; ++nt) {
    int col = nt * 16 + m;
    float bv = b1[col];
    float wv = W2[col];
#pragma unroll
    for (int r = 0; r < 8; ++r) {
      float h = acc[nt][r] + bv;
      h = h > 0.f ? h : 0.f;
      part[r] += h * wv;
    }
  }
  // reduce over the 16 lanes sharing each row group (masks < 16 keep hi fixed)
#pragma unroll
  for (int mask = 8; mask >= 1; mask >>= 1) {
#pragma unroll
    for (int r = 0; r < 8; ++r) part[r] += __shfl_xor(part[r], mask, 32);
  }
  if (m == 0) {
    float bo = b2[0];
#pragma unroll
    for (int r = 0; r < 8; ++r) {
      int row = rowbase + r + 8 * hi;
      if (row < E) out[row] = part[r] + bo;
    }
  }
}

extern "C" void kernel_launch(void* const* d_in, const int* in_sizes, int n_in,
                              void* d_out, int out_size, void* d_ws, size_t ws_size,
                              hipStream_t stream) {
  const int*   etype = (const int*)d_in[0];
  const int*   ab    = (const int*)d_in[1];
  const int*   bc    = (const int*)d_in[2];
  const int*   ac    = (const int*)d_in[3];
  const float* fact  = (const float*)d_in[4];
  const float* W1    = (const float*)d_in[5];
  const float* b1    = (const float*)d_in[6];
  const float* W2    = (const float*)d_in[7];
  const float* b2    = (const float*)d_in[8];
  const float* mW1   = (const float*)d_in[9];
  const float* mb1   = (const float*)d_in[10];
  const float* mW2   = (const float*)d_in[11];
  const float* mb2   = (const float*)d_in[12];
  float* outp = (float*)d_out;

  int E = in_sizes[0];
  int T = in_sizes[1];
  size_t feat = (size_t)E * D1;

  float* ws  = (float*)d_ws;
  float* x0  = ws;
  float* x   = ws + feat;
  float* agg = ws + 2 * feat;

  dim3 blk(256);
  int gfeat = (int)((feat + 255) / 256);
  int gtri  = (int)(((size_t)T * 64 + 255) / 256);
  int ggemm = (E + 127) / 128;

  gather_kernel<<<gfeat, blk, 0, stream>>>(fact, etype, x0, E);

  // layer 1
  zero_kernel<<<gfeat, blk, 0, stream>>>(agg, (int)feat);
  tri_kernel<<<gtri, blk, 0, stream>>>(x0, ab, bc, ac, agg, T);
  layer_gemm<<<ggemm, blk, 0, stream>>>(agg, W1, b1, x, E);

  // layer 2
  zero_kernel<<<gfeat, blk, 0, stream>>>(agg, (int)feat);
  tri_kernel<<<gtri, blk, 0, stream>>>(x, ab, bc, ac, agg, T);
  layer_gemm<<<ggemm, blk, 0, stream>>>(agg, W2, b2, x, E);

  // fused MLP head
  final_mlp<<<ggemm, blk, 0, stream>>>(x, x0, mW1, mb1, mW2, mb2, outp, E);
}
